// FlowGen_53541062312436
// MI455X (gfx1250) — matrix operand order repaired and checked by reading, not compiled, not run on hardware
//
#include <hip/hip_runtime.h>
#include <hip/hip_bf16.h>
#include <math.h>

typedef _Float16 v16h __attribute__((ext_vector_type(16)));
typedef _Float16 v8h  __attribute__((ext_vector_type(8)));
typedef float    v8f  __attribute__((ext_vector_type(8)));

#define NLAYERS 45
#define TILE    128   // samples per workgroup
#define XSTR    18    // padded row stride for 17-dim state
#define NOSTR   224   // padded stride of MLP output tile (max out = 207)

// layer type: 0=actnorm 1=lu 2=affine 3=rq
__device__ __constant__ int d_LT[NLAYERS] = {
  0,1,2,
  0,1,2,
  1,2,1,3,1,3,1,3,
  1,2,1,3,1,3,1,3,
  1,2,1,3,1,3,1,3,
  1,2,1,3,1,3,1,3,
  1,3,1,3,1,3,
  1};
__device__ __constant__ int d_LM[NLAYERS] = {
  0,0,0,
  0,0,1,
  0,0,0,0,0,0,0,0,
  0,1,0,1,0,1,0,1,
  0,0,0,0,0,0,0,0,
  0,1,0,1,0,1,0,1,
  0,0,0,1,0,0,
  0};

struct KP {
  const float* xy;
  const int*   labels;
  const float* emb;
  float*       out;
  int          N;
  const float* p[NLAYERS][6];
};

// ---------------- LDS layout (dynamic shared) ----------------
#define OFF_XS   0                         // float [128][18]  9216 B
#define OFF_XT   9216                      // float [128][18]  9216 B
#define OFF_CTX  18432                     // half  [128][8]   2048 B
#define OFF_AIN  20480                     // half  [128][32]  8192 B
#define OFF_WL   28672                     // half  [208][64] 26624 B
#define OFF_H1   55296                     // half  [128][64] 16384 B
#define OFF_H2   71680                     // half  [128][64] 16384 B
#define OFF_NOUT 88064                     // half  [128][224] 57344 B
#define OFF_ML   145408                    // float [17][20]   1360 B
#define OFF_UDD  146768                    // float [17]  (pad 80)
#define OFF_LAD  146848                    // float [128]       512 B
#define OFF_MISC 147360                    // float scratch
#define SMEM_BYTES 147392

__device__ __forceinline__ float softplus_f(float x) {
  return (x > 20.f) ? x : log1pf(expf(x));
}

// -------- WMMA fragment loads (CDNA5 documented VGPR layouts) --------
// A (16x32 f16): lanes 0-15 hold K {0..7,16..23}, lanes 16-31 K {8..15,24..31}
__device__ __forceinline__ v16h loadA(const _Float16* A, int stride, int mbase,
                                      int k0, int lane) {
  int m   = mbase + (lane & 15);
  int sel = lane >> 4;
  const _Float16* rp = A + m * stride + k0 + sel * 8;
  v8h lo = *(const v8h*)rp;
  v8h hi = *(const v8h*)(rp + 16);
  v16h r;
#pragma unroll
  for (int i = 0; i < 8; ++i) { r[i] = lo[i]; r[i + 8] = hi[i]; }
  return r;
}
// B (32x16 f16): lane n = col; lanes 0-15 K {0..7,16..23}, lanes 16-31 K {8..15,24..31}.
// B[k][n] = W[n][k] (W row-major [out][K]) -> same K order as A.
__device__ __forceinline__ v16h loadB(const _Float16* W, int wstride, int nbase,
                                      int k0, int lane) {
  int n  = nbase + (lane & 15);
  int ks = lane >> 4;
  const _Float16* rp = W + n * wstride + k0 + ks * 8;
  v8h lo = *(const v8h*)rp;
  v8h hi = *(const v8h*)(rp + 16);
  v16h r;
#pragma unroll
  for (int i = 0; i < 8; ++i) { r[i] = lo[i]; r[i + 8] = hi[i]; }
  return r;
}

// One wave computes a 16-row M-tile of  O = act(A @ W^T + b)  with f32 accum.
__device__ __forceinline__ void gemm_tile(const _Float16* A, int astride,
                                          const _Float16* W, int wstride,
                                          const float* bias, int outReal,
                                          int OUTP, int K, _Float16* O,
                                          int ostride, bool relu, int lane,
                                          int mbase) {
  v16h af[2];
  const int nk = K >> 5;
#pragma unroll 2
  for (int kk = 0; kk < nk; ++kk) af[kk] = loadA(A, astride, mbase, kk * 32, lane);
  const int n   = lane & 15;
  const int sel = lane >> 4;
  for (int nb = 0; nb < OUTP; nb += 16) {
    v8f c = {};
#pragma unroll 2
    for (int kk = 0; kk < nk; ++kk) {
      v16h bf = loadB(W, wstride, nb, kk * 32, lane);
      c = __builtin_amdgcn_wmma_f32_16x16x32_f16(false, af[kk], false, bf,
                                                 (short)0, c, false, false);
    }
    int ng   = nb + n;
    float bv = (ng < outReal) ? bias[ng] : 0.f;
#pragma unroll
    for (int v = 0; v < 8; ++v) {
      float x = c[v] + bv;
      if (relu) x = fmaxf(x, 0.f);
      O[(mbase + v + sel * 8) * ostride + ng] = (_Float16)x;
    }
  }
}

// Cooperative f32 -> f16 weight staging with zero padding.
__device__ __forceinline__ void loadW(_Float16* dst, const float* src,
                                      int rowsPad, int rowsReal, int colsReal,
                                      int cshift, int tid) {
  const int colsPad = 1 << cshift;
  for (int e = tid; e < rowsPad * colsPad; e += 256) {
    int r = e >> cshift;
    int c = e & (colsPad - 1);
    float v = (r < rowsReal && c < colsReal) ? src[r * colsReal + c] : 0.f;
    dst[e] = (_Float16)v;
  }
}

__global__ __launch_bounds__(256) void flowgen_kernel(KP kp) {
  extern __shared__ char smem[];
  float*    xs   = (float*)(smem + OFF_XS);
  float*    xt   = (float*)(smem + OFF_XT);
  _Float16* ctx  = (_Float16*)(smem + OFF_CTX);
  _Float16* Ain  = (_Float16*)(smem + OFF_AIN);
  _Float16* Wl   = (_Float16*)(smem + OFF_WL);
  _Float16* Hb1  = (_Float16*)(smem + OFF_H1);
  _Float16* Hb2  = (_Float16*)(smem + OFF_H2);
  _Float16* Nout = (_Float16*)(smem + OFF_NOUT);
  float*    Ml   = (float*)(smem + OFF_ML);
  float*    udd  = (float*)(smem + OFF_UDD);
  float*    lads = (float*)(smem + OFF_LAD);
  float*    misc = (float*)(smem + OFF_MISC);

  const int tid   = threadIdx.x;
  const int lane  = tid & 31;
  const int mbase = (tid >> 5) << 4;   // 8 waves x 16 rows = 128-row tile
  const int s0    = blockIdx.x * TILE;

  // ---- load state tile, embeddings, init log-det ----
  for (int e = tid; e < TILE * 17; e += 256) {
    int s = e / 17, d = e - 17 * s;
    xs[s * XSTR + d] = kp.xy[(size_t)(s0 + s) * 17 + d];
  }
  for (int e = tid; e < TILE * 8; e += 256) {
    int s = e >> 3, j = e & 7;
    ctx[s * 8 + j] = (_Float16)kp.emb[kp.labels[s0 + s] * 8 + j];
  }
  if (tid < TILE) lads[tid] = 0.f;
  __syncthreads();

  for (int l = 0; l < NLAYERS; ++l) {
    const int lt = d_LT[l];
    const int mi = d_LM[l];
    const float* const* P = kp.p[l];

    if (lt == 0) {  // -------- actnorm --------
      const float* ls = P[0];
      const float* sh = P[1];
      if (tid == 0) {
        float sm = 0.f;
        for (int d = 0; d < 17; ++d) sm += ls[d];
        misc[0] = sm;
      }
      for (int e = tid; e < TILE * 17; e += 256) {
        int s = e / 17, d = e - 17 * s;
        xs[s * XSTR + d] = xs[s * XSTR + d] * expf(ls[d]) + sh[d];
      }
      __syncthreads();
      if (tid < TILE) lads[tid] += misc[0];
      __syncthreads();

    } else if (lt == 1) {  // -------- LU coupling --------
      const float *Lg = P[0], *Ug = P[1], *bg = P[2], *ug = P[3];
      if (tid < 17) udd[tid] = 0.001f + softplus_f(ug[tid]);
      __syncthreads();
      if (tid == 0) {
        float sm = 0.f;
        for (int k = 0; k < 17; ++k) sm += logf(udd[k]);
        misc[0] = sm;
      }
      for (int e = tid; e < 289; e += 256) {  // M = L @ U
        int d = e / 17, k = e - 17 * d;
        int mn = d < k ? d : k;
        float acc = 0.f;
        for (int j = 0; j <= mn; ++j) {
          float lv = (j < d) ? Lg[d * 17 + j] : 1.0f;
          float uv = (j < k) ? Ug[j * 17 + k] : udd[k];
          acc += lv * uv;
        }
        Ml[d * 20 + k] = acc;
      }
      __syncthreads();
      for (int e = tid; e < TILE * 17; e += 256) {  // x' = x @ M^T + bias
        int s = e / 17, d = e - 17 * s;
        const float* xr = &xs[s * XSTR];
        float acc = bg[d];
#pragma unroll
        for (int k = 0; k < 17; ++k) acc += Ml[d * 20 + k] * xr[k];
        xt[s * XSTR + d] = acc;
      }
      __syncthreads();
      for (int e = tid; e < TILE * 17; e += 256) {
        int s = e / 17, d = e - 17 * s;
        xs[s * XSTR + d] = xt[s * XSTR + d];
      }
      if (tid < TILE) lads[tid] += misc[0];
      __syncthreads();

    } else {  // -------- coupling with MLP (affine=2 / rq=3) --------
      const int ni      = 9 - mi;          // identity features
      const int nt      = 8 + mi;          // transformed features
      const int inReal  = ni + 8;          // + EMB
      const int outReal = (lt == 2) ? 2 * nt : nt * 23;
      const int OUTP    = (outReal + 15) & ~15;

      // build A = [x[:,ids], ctx] padded to K=32
      for (int e = tid; e < TILE * 32; e += 256) {
        int s = e >> 5, c = e & 31;
        _Float16 v = (_Float16)0.f;
        if (c < ni)            v = (_Float16)xs[s * XSTR + 2 * c + mi];
        else if (c < ni + 8)   v = ctx[s * 8 + (c - ni)];
        Ain[e] = v;
      }
      loadW(Wl, P[0], 64, 64, inReal, 5, tid);
      __syncthreads();
      gemm_tile(Ain, 32, Wl, 32, P[3], 64, 64, 32, Hb1, 64, true, lane, mbase);
      __syncthreads();
      loadW(Wl, P[1], 64, 64, 64, 6, tid);
      __syncthreads();
      gemm_tile(Hb1, 64, Wl, 64, P[4], 64, 64, 64, Hb2, 64, true, lane, mbase);
      __syncthreads();
      loadW(Wl, P[2], OUTP, outReal, 64, 6, tid);
      __syncthreads();
      gemm_tile(Hb2, 64, Wl, 64, P[5], outReal, OUTP, 64, Nout, NOSTR, false,
                lane, mbase);
      __syncthreads();

      // -------- elementwise transform --------
      if (lt == 2) {  // affine
        for (int e = tid; e < TILE * nt; e += 256) {
          int s, t;
          if (nt == 8) { s = e >> 3; t = e & 7; }
          else         { s = e / 9;  t = e - 9 * s; }
          int col = 2 * t + 1 - mi;
          float shift = (float)Nout[s * NOSTR + t];
          float z     = (float)Nout[s * NOSTR + nt + t] + 2.0f;
          float scale = 1.0f / (1.0f + expf(-z)) + 0.001f;
          xs[s * XSTR + col] = xs[s * XSTR + col] * scale + shift;
          atomicAdd(&lads[s], logf(scale));
        }
      } else {  // rational-quadratic spline
        for (int e = tid; e < TILE * nt; e += 256) {
          int s, t;
          if (nt == 8) { s = e >> 3; t = e & 7; }
          else         { s = e / 9;  t = e - 9 * s; }
          const _Float16* row = Nout + s * NOSTR + t * 23;
          float uw[8], uh[8], udm[7];
#pragma unroll
          for (int k = 0; k < 8; ++k) { uw[k] = (float)row[k]; uh[k] = (float)row[8 + k]; }
#pragma unroll
          for (int k = 0; k < 7; ++k) udm[k] = (float)row[16 + k];

          // softmax -> bin fractions (sum stays exactly 1 after min-bin mix)
          float mw = uw[0], mh = uh[0];
#pragma unroll
          for (int k = 1; k < 8; ++k) { mw = fmaxf(mw, uw[k]); mh = fmaxf(mh, uh[k]); }
          float sw = 0.f, sh = 0.f;
          float ew[8], eh[8];
#pragma unroll
          for (int k = 0; k < 8; ++k) {
            ew[k] = expf(uw[k] - mw); sw += ew[k];
            eh[k] = expf(uh[k] - mh); sh += eh[k];
          }
          float rw = 0.992f / sw, rh = 0.992f / sh;

          int col = 2 * t + 1 - mi;
          float x0 = xs[s * XSTR + col];
          bool inside = fabsf(x0) <= 3.0f;
          float xc = fminf(fmaxf(x0, -3.0f), 3.0f);

          // branchless bin search with fused cumsum (registers only)
          float cwp = -3.f, chp = -3.f, dprev = 1.0f, accw = 0.f, acch = 0.f;
          float in_cw = -3.f, in_w = 1.f, in_ch = -3.f, in_h = 1.f, d0 = 1.f, d1 = 1.f;
#pragma unroll
          for (int k = 0; k < 8; ++k) {
            accw += 0.001f + rw * ew[k];
            acch += 0.001f + rh * eh[k];
            float cwn = (k == 7) ? 3.0f : (-3.0f + 6.0f * accw);
            float chn = (k == 7) ? 3.0f : (-3.0f + 6.0f * acch);
            float dn  = (k == 7) ? 1.0f : (0.001f + softplus_f(udm[k]));
            if (xc >= cwp) {
              in_cw = cwp; in_w = cwn - cwp;
              in_ch = chp; in_h = chn - chp;
              d0 = dprev;  d1 = dn;
            }
            cwp = cwn; chp = chn; dprev = dn;
          }

          float theta = (xc - in_cw) / in_w;
          float omt   = 1.0f - theta;
          float t1m   = theta * omt;
          float delta = in_h / in_w;
          float th2   = theta * theta;
          float num   = in_h * (delta * th2 + d0 * t1m);
          float den   = delta + (d0 + d1 - 2.0f * delta) * t1m;
          float y     = in_ch + num / den;
          float dnum  = delta * delta * (d1 * th2 + 2.0f * delta * t1m + d0 * omt * omt);
          float ld    = logf(dnum) - 2.0f * logf(den);

          xs[s * XSTR + col] = inside ? y : x0;
          atomicAdd(&lads[s], inside ? ld : 0.0f);
        }
      }
      __syncthreads();
    }
  }

  // ---- write outputs: x (N*17 floats) then lad (N floats) ----
  for (int e = tid; e < TILE * 17; e += 256) {
    int s = e / 17, d = e - 17 * s;
    kp.out[(size_t)(s0 + s) * 17 + d] = xs[s * XSTR + d];
  }
  if (tid < TILE) kp.out[(size_t)kp.N * 17 + s0 + tid] = lads[tid];
}

extern "C" void kernel_launch(void* const* d_in, const int* in_sizes, int n_in,
                              void* d_out, int out_size, void* d_ws, size_t ws_size,
                              hipStream_t stream) {
  static const int h_LT[NLAYERS] = {
    0,1,2,
    0,1,2,
    1,2,1,3,1,3,1,3,
    1,2,1,3,1,3,1,3,
    1,2,1,3,1,3,1,3,
    1,2,1,3,1,3,1,3,
    1,3,1,3,1,3,
    1};

  KP kp;
  kp.xy     = (const float*)d_in[0];
  kp.labels = (const int*)d_in[1];
  int idx = 2;
  for (int l = 0; l < NLAYERS; ++l) {
    int ns = (h_LT[l] == 0) ? 2 : (h_LT[l] == 1) ? 4 : 6;
    for (int j = 0; j < 6; ++j)
      kp.p[l][j] = (j < ns) ? (const float*)d_in[idx + j] : nullptr;
    idx += ns;
  }
  kp.emb = (const float*)d_in[idx];  // embedding is the last input
  kp.out = (float*)d_out;
  kp.N   = in_sizes[0] / 17;

  const int blocks = kp.N / TILE;  // 262144 / 128 = 2048
  flowgen_kernel<<<blocks, 256, SMEM_BYTES, stream>>>(kp);
}
